// QuantLinearW4Grouped_26113401159877
// MI455X (gfx1250) — matrix operand order repaired and checked by reading, not compiled, not run on hardware
//
#include <hip/hip_runtime.h>
#include <hip/hip_bf16.h>

// Problem constants (match reference)
#define IN_F     4096
#define OUT_F    11008
#define GROUP    128
#define N_GROUPS (IN_F / GROUP)
#define TOKENS   4096

// Tiling
#define BM 128          // token rows per block
#define BN 128          // out-features per block
#define BK 128          // K chunk == quant group size
#define BKP (BK + 8)    // padded LDS row (272B stride -> rotating banks)
#define NCHUNKS (IN_F / BK)

typedef __attribute__((ext_vector_type(16))) __bf16 bf16x16;
typedef __attribute__((ext_vector_type(8)))  float  f32x8;

union Frag16 {
    bf16x16 v;
    uint4   u[2];
};

#if __has_builtin(__builtin_amdgcn_global_load_async_to_lds_b128)
#define HAVE_ASYNC 1
// Builtin expects pointers to 16-byte int vectors (per hipcc diagnostic):
//   param0: v4i in global/device address space
//   param1: v4i in LDS address space
typedef int v4i_async __attribute__((__vector_size__(16)));
typedef __attribute__((address_space(1))) v4i_async g_v4i;
typedef __attribute__((address_space(3))) v4i_async l_v4i;
#else
#define HAVE_ASYNC 0
#endif

#if __has_builtin(__builtin_amdgcn_s_wait_asynccnt)
#define WAIT_ASYNC(N) __builtin_amdgcn_s_wait_asynccnt(N)
#else
#define WAIT_ASYNC(N) asm volatile("s_wait_asynccnt %0" ::"i"(N) : "memory")
#endif

__device__ __forceinline__ unsigned int f32_to_bf16_rne(float f) {
    union { float f; unsigned int u; } cv;
    cv.f = f;
    unsigned int r = cv.u + 0x7FFFu + ((cv.u >> 16) & 1u);
    return r >> 16;
}

__device__ __forceinline__ unsigned int pack_bf16_rne(float lo, float hi) {
    return f32_to_bf16_rne(lo) | (f32_to_bf16_rne(hi) << 16);
}

// Exact pack for small integers (|v| <= 127): bf16 truncation is lossless.
__device__ __forceinline__ unsigned int pack_bf16_trunc(float lo, float hi) {
    union { float f; unsigned int u; } a, b;
    a.f = lo; b.f = hi;
    return (a.u >> 16) | (b.u & 0xFFFF0000u);
}

// sign-extended byte b of word w -> float (v_bfe_i32 + v_cvt_f32_i32)
__device__ __forceinline__ float b2f(int w, int b) {
    return (float)((signed char)((w >> (8 * b)) & 0xFF));
}

// ---------------------------------------------------------------------------
// Pre-pass: x f32 -> bf16 (RNE) once, instead of 86x inside the GEMM.
// ---------------------------------------------------------------------------
__global__ __launch_bounds__(256) void xcvt_kernel(const float* __restrict__ x,
                                                   unsigned short* __restrict__ xb) {
    size_t i = (size_t)blockIdx.x * 256 + threadIdx.x;   // 8 floats per thread
    const float4* xp = (const float4*)x + 2 * i;
    float4 f0 = xp[0], f1 = xp[1];
    uint4 p;
    p.x = pack_bf16_rne(f0.x, f0.y);
    p.y = pack_bf16_rne(f0.z, f0.w);
    p.z = pack_bf16_rne(f1.x, f1.y);
    p.w = pack_bf16_rne(f1.z, f1.w);
    ((uint4*)xb)[i] = p;
}

// ---------------------------------------------------------------------------
// Main GEMM. XBF16: activations come pre-converted (bf16) from workspace.
// Weights are stored in LDS as raw q (exact in bf16); the per-(n,group) scale
// is applied AFTER the group's WMMAs: acc += gacc * scale.
// ---------------------------------------------------------------------------
template <bool XBF16>
__global__ __launch_bounds__(256) void w4a16_wmma_gemm(
    const float*          __restrict__ x,       // [TOKENS, IN_F] f32 (fallback)
    const unsigned short* __restrict__ xb,      // [TOKENS, IN_F] bf16 bits
    const signed char*    __restrict__ w_q,     // [OUT_F, IN_F] int8 (int4 range)
    const float*          __restrict__ scales,  // [OUT_F, N_GROUPS]
    const float*          __restrict__ bias,    // [OUT_F]
    float*                __restrict__ out)     // [TOKENS, OUT_F]
{
    __shared__ unsigned short sA[2][BM * BKP];  // activations bf16 (double buf)
    __shared__ unsigned short sB[BN * BKP];     // raw q as bf16

    const int tid  = threadIdx.x;
    const int lane = tid & 31;
    const int wave = tid >> 5;
    const int wm   = wave >> 1;     // 0..3
    const int wn   = wave & 1;      // 0..1
    const int lr   = lane & 15;
    const int hi   = lane >> 4;

    const int tileN = blockIdx.x * BN;
    const int tileM = blockIdx.y * BM;

    const int rowS = tid >> 1;          // 0..127
    const int half = tid & 1;           // 64-element halves

    const signed char*    wg   = w_q + (size_t)(tileN + rowS) * IN_F + half * 64;
    const float*          xg32 = x  + (size_t)(tileM + rowS) * IN_F + half * 64;
    const unsigned short* xg16 = xb + (size_t)(tileM + rowS) * IN_F + half * 64;

    const int aLdsOff = rowS * BKP + half * 64;   // element offset in sA buffer
    unsigned short* sBw = &sB[rowS * BKP + half * 64];

    int4 wReg[4];

#define GLOADB(KC)                                                         \
    do {                                                                   \
        const signed char* wp = wg + (size_t)(KC) * BK;                    \
        _Pragma("unroll")                                                  \
        for (int j = 0; j < 4; ++j)                                        \
            wReg[j] = *(const int4*)(wp + j * 16);                         \
    } while (0)

#if HAVE_ASYNC
#define ISSUE_A(KC, BUF)                                                   \
    do {                                                                   \
        const unsigned short* gp = xg16 + (size_t)(KC) * BK;               \
        unsigned short* lp = &sA[(BUF)][aLdsOff];                          \
        _Pragma("unroll")                                                  \
        for (int j = 0; j < 8; ++j)                                        \
            __builtin_amdgcn_global_load_async_to_lds_b128(                \
                (g_v4i*)(gp + j * 8), (l_v4i*)(lp + j * 8), 0, 0);         \
    } while (0)
#endif

    f32x8 acc[2][4];
#pragma unroll
    for (int i = 0; i < 2; ++i)
#pragma unroll
        for (int j = 0; j < 4; ++j)
            acc[i][j] = (f32x8)(0.0f);

    const unsigned short* bBase = &sB[(wn * 64 + lr) * BKP];

    // ---- prologue ----
    GLOADB(0);
    if constexpr (XBF16) {
#if HAVE_ASYNC
        ISSUE_A(0, 0);
#endif
    }

    for (int kc = 0; kc < NCHUNKS; ++kc) {
        __syncthreads();   // previous chunk's LDS reads complete

        // ---- weight regs -> LDS: raw q -> bf16 (exact, no scale) ----
#pragma unroll
        for (int j = 0; j < 4; ++j) {
            int4 w = wReg[j];
            uint4 q0, q1;
            q0.x = pack_bf16_trunc(b2f(w.x, 0), b2f(w.x, 1));
            q0.y = pack_bf16_trunc(b2f(w.x, 2), b2f(w.x, 3));
            q0.z = pack_bf16_trunc(b2f(w.y, 0), b2f(w.y, 1));
            q0.w = pack_bf16_trunc(b2f(w.y, 2), b2f(w.y, 3));
            q1.x = pack_bf16_trunc(b2f(w.z, 0), b2f(w.z, 1));
            q1.y = pack_bf16_trunc(b2f(w.z, 2), b2f(w.z, 3));
            q1.z = pack_bf16_trunc(b2f(w.w, 0), b2f(w.w, 1));
            q1.w = pack_bf16_trunc(b2f(w.w, 2), b2f(w.w, 3));
            *(uint4*)(sBw + j * 16)     = q0;
            *(uint4*)(sBw + j * 16 + 8) = q1;
        }

        // ---- activation tile provisioning ----
        int abuf = 0;
        if constexpr (XBF16) {
#if HAVE_ASYNC
            // double-buffered async copy; wait for this chunk's batch only
            if (kc + 1 < NCHUNKS) {
                ISSUE_A(kc + 1, (kc + 1) & 1);
                WAIT_ASYNC(8);
            } else {
                WAIT_ASYNC(0);
            }
            abuf = kc & 1;
#else
            { // plain LDS copy of pre-converted bf16
                const unsigned short* gp = xg16 + (size_t)kc * BK;
                unsigned short* lp = &sA[0][aLdsOff];
#pragma unroll
                for (int j = 0; j < 8; ++j)
                    *(uint4*)(lp + j * 8) = *(const uint4*)(gp + j * 8);
            }
#endif
        } else {
            // fallback: load f32 and convert inline
            const float* gp = xg32 + (size_t)kc * BK;
            unsigned short* lp = &sA[0][aLdsOff];
#pragma unroll
            for (int j = 0; j < 8; ++j) {
                float4 f0 = *(const float4*)(gp + j * 8);
                float4 f1 = *(const float4*)(gp + j * 8 + 4);
                uint4 p;
                p.x = pack_bf16_rne(f0.x, f0.y);
                p.y = pack_bf16_rne(f0.z, f0.w);
                p.z = pack_bf16_rne(f1.x, f1.y);
                p.w = pack_bf16_rne(f1.z, f1.w);
                *(uint4*)(lp + j * 8) = p;
            }
        }

        __syncthreads();   // LDS tiles ready (each wave waited its ASYNCcnt)

        // ---- overlap next chunk's global traffic with the WMMAs ----
        if (kc + 1 < NCHUNKS)
            GLOADB(kc + 1);
        if (kc + 2 < NCHUNKS) {
            __builtin_prefetch(wg + (size_t)(kc + 2) * BK, 0, 0);
            if constexpr (XBF16)
                __builtin_prefetch(xg16 + (size_t)(kc + 2) * BK, 0, 0);
            else
                __builtin_prefetch(xg32 + (size_t)(kc + 2) * BK, 0, 0);
        }

        // per-(n, group) scales for the merge below
        float smul[4];
#pragma unroll
        for (int j = 0; j < 4; ++j)
            smul[j] = scales[(size_t)(tileN + wn * 64 + j * 16 + lr) * N_GROUPS + kc];

        // ---- WMMA: accumulate this group's raw-q product into gacc ----
        const unsigned short* aBase = &sA[abuf][(wm * 32 + lr) * BKP];
        f32x8 gacc[2][4];
#pragma unroll
        for (int i = 0; i < 2; ++i)
#pragma unroll
            for (int j = 0; j < 4; ++j)
                gacc[i][j] = (f32x8)(0.0f);

#pragma unroll
        for (int kk = 0; kk < 4; ++kk) {
            const int ka = kk * 32 + hi * 8;    // A: K..K+7 | K+16..K+23 split
            const int kb = kk * 32 + hi * 8;    // B: K..K+7 | K+16..K+23 split

            Frag16 a[2];
#pragma unroll
            for (int i = 0; i < 2; ++i) {
                const unsigned short* p = aBase + i * 16 * BKP + ka;
                a[i].u[0] = *(const uint4*)(p);
                a[i].u[1] = *(const uint4*)(p + 16);
            }
            Frag16 b[4];
#pragma unroll
            for (int j = 0; j < 4; ++j) {
                const unsigned short* p = bBase + j * 16 * BKP + kb;
                b[j].u[0] = *(const uint4*)(p);
                b[j].u[1] = *(const uint4*)(p + 16);
            }
#pragma unroll
            for (int i = 0; i < 2; ++i)
#pragma unroll
                for (int j = 0; j < 4; ++j)
                    gacc[i][j] = __builtin_amdgcn_wmma_f32_16x16x32_bf16(
                        false, a[i].v, false, b[j].v,
                        (short)0, gacc[i][j], false, false);
        }

        // ---- apply group scale in f32 and accumulate ----
#pragma unroll
        for (int i = 0; i < 2; ++i)
#pragma unroll
            for (int j = 0; j < 4; ++j)
#pragma unroll
                for (int r = 0; r < 8; ++r)
                    acc[i][j][r] += gacc[i][j][r] * smul[j];
    }

    // ---- epilogue: C/D layout -> global, add bias ----
#pragma unroll
    for (int i = 0; i < 2; ++i) {
        const int m0 = tileM + wm * 32 + i * 16 + hi * 8;
#pragma unroll
        for (int j = 0; j < 4; ++j) {
            const int n  = tileN + wn * 64 + j * 16 + lr;
            const float bv = bias[n];
            float* op = out + (size_t)m0 * OUT_F + n;
#pragma unroll
            for (int r = 0; r < 8; ++r)
                op[(size_t)r * OUT_F] = acc[i][j][r] + bv;
        }
    }
#undef GLOADB
#if HAVE_ASYNC
#undef ISSUE_A
#endif
}

extern "C" void kernel_launch(void* const* d_in, const int* in_sizes, int n_in,
                              void* d_out, int out_size, void* d_ws, size_t ws_size,
                              hipStream_t stream) {
    const float*       x      = (const float*)d_in[0];
    const signed char* w_q    = (const signed char*)d_in[1];
    const float*       scales = (const float*)d_in[2];
    const float*       bias   = (const float*)d_in[3];
    float*             out    = (float*)d_out;

    dim3 grid(OUT_F / BN, TOKENS / BM);  // 86 x 32
    dim3 block(256);

    const size_t need = (size_t)TOKENS * IN_F * sizeof(unsigned short); // 32 MB
    if (ws_size >= need) {
        unsigned short* xb = (unsigned short*)d_ws;
        const int nconv = (int)(((size_t)TOKENS * IN_F) / (256 * 8));   // 8192
        xcvt_kernel<<<nconv, 256, 0, stream>>>(x, xb);
        w4a16_wmma_gemm<true><<<grid, block, 0, stream>>>(x, xb, w_q, scales, bias, out);
    } else {
        w4a16_wmma_gemm<false><<<grid, block, 0, stream>>>(x, (const unsigned short*)nullptr,
                                                           w_q, scales, bias, out);
    }
}
